// Geo_FFNO_2052994368057
// MI455X (gfx1250) — compile-verified, matrix operand order checked by reading, not run on hardware
//
#include <hip/hip_runtime.h>
#include <hip/hip_bf16.h>

// FFNO block for MI455X (gfx1250, wave32, WMMA + TDM).
// Heavy math: v_wmma_f32_16x16x32_f16, K-contiguous fragments (2x b128 each).
// f16 LDS staging done by the Tensor Data Mover (tensor_load_to_lds).

typedef __attribute__((ext_vector_type(16))) _Float16 v16h;
typedef __attribute__((ext_vector_type(8)))  _Float16 v8h;
typedef __attribute__((ext_vector_type(8)))  float    v8f;
typedef unsigned int uint32x4 __attribute__((ext_vector_type(4)));
typedef int          int32x4  __attribute__((ext_vector_type(4)));
typedef int          int32x8  __attribute__((ext_vector_type(8)));

#define LN_EPS 1e-5f
#define PI_F 3.14159265358979323846f

#define Bn 16
#define Mn 128
#define Nn 128
#define Dn 128
#define NMODE 16
#define Hn 512
#define ROWS (Bn*Mn)   // 2048 (== Bn*Nn)
#define KC 32          // 16 modes * {re,im}
#define XLD 136        // padded LDS ld (halves): 272 B rows, 16B-aligned

// ---- Tensor Data Mover: 1-D async copy global->LDS (D# per ISA 08 s8.3/8.4)
// bytes must be a multiple of 8; tile_dim0 (16-bit) = bytes/8 units.
__device__ inline void tdm_copy_1d(const void* gsrc, void* lds_dst, unsigned units8) {
#if __has_builtin(__builtin_amdgcn_tensor_load_to_lds)
  unsigned long long ga = (unsigned long long)gsrc;
  unsigned lds = (unsigned)(unsigned long long)lds_dst; // low 32 bits = LDS offset
  uint32x4 g0;
  g0.x = 1u;                                         // count=1 valid user D#
  g0.y = lds;                                        // lds_addr (bytes)
  g0.z = (unsigned)ga;                               // global_addr[31:0]
  g0.w = ((unsigned)(ga >> 32) & 0x1FFFFFFu) | (2u << 30); // addr[56:32] | type=2
  int32x8 g1;
  g1[0] = (int)(3u << 16);                           // data_size=3 (8 bytes)
  g1[1] = (int)((units8 & 0xFFFFu) << 16);           // tensor_dim0[15:0] @ bits 63:48
  g1[2] = (int)((units8 >> 16) & 0xFFFFu);           // tensor_dim0[31:16]; dim1=0
  g1[3] = (int)((units8 & 0xFFFFu) << 16);           // tile_dim0 @ bits 127:112
  g1[4] = 0;                                         // tile_dim1/2 = 0 (unused)
  g1[5] = (int)units8;                               // tensor_dim0_stride[31:0]
  g1[6] = 0;
  g1[7] = 0;
  int32x4 gz = {0, 0, 0, 0};
#if __clang_major__ >= 23
  int32x8 gz8 = {0, 0, 0, 0, 0, 0, 0, 0};
  __builtin_amdgcn_tensor_load_to_lds(g0, g1, gz, gz, gz8, 0);
#else
  __builtin_amdgcn_tensor_load_to_lds(g0, g1, gz, gz, 0);
#endif
#if __has_builtin(__builtin_amdgcn_s_wait_tensorcnt)
  __builtin_amdgcn_s_wait_tensorcnt((short)0);       // s_wait_tensorcnt 0
#endif
#endif
}
#if __has_builtin(__builtin_amdgcn_tensor_load_to_lds)
#define HAVE_TDM 1
#else
#define HAVE_TDM 0
#endif

// ---- wave32 WMMA fragment loader (K-contiguous source) ----
// lane < 16:  row = lane,    K {0..7, 16..23}
// lane >= 16: row = lane-16, K {8..15, 24..31}
// => two 16-byte loads per lane at byte offsets {kb*2, 32+kb*2}, kb in {0,8}
__device__ inline v16h frag_k(const _Float16* base, int ld) {
  const int lane = threadIdx.x & 31;
  const int row  = lane & 15;
  const int kb   = (lane >> 4) << 3;
  const _Float16* p = base + row * ld + kb;
  v8h lo = *(const v8h*)(p);       // ds_load_b128 / global_load_b128
  v8h hi = *(const v8h*)(p + 16);
  return __builtin_shufflevector(lo, hi, 0,1,2,3,4,5,6,7,8,9,10,11,12,13,14,15);
}
__device__ inline v8f wmma_f16(v16h a, v16h b, v8f c) {
  return __builtin_amdgcn_wmma_f32_16x16x32_f16(false, a, false, b, (short)0, c,
                                                false, false);
}

// ---- K0a: DFT tables (f16), ortho norm + irfft alpha folded in ----
__global__ void k_tables(_Float16* __restrict__ Ftab, _Float16* __restrict__ Tinv) {
  const float inv = 1.0f / sqrtf((float)Nn);
  for (int t = threadIdx.x; t < KC * Nn; t += blockDim.x) {
    int kc = t / Nn, n = t % Nn;
    int k = kc & 15;
    float ang = 2.0f * PI_F * (float)(k * n) / (float)Nn;
    float v = (kc < 16) ? cosf(ang) : -sinf(ang);
    Ftab[kc * Nn + n] = (_Float16)(v * inv);           // [32 kc][128 n], K-contig
  }
  for (int t = threadIdx.x; t < Nn * KC; t += blockDim.x) {
    int n = t / KC, q = t % KC;
    int k = q & 15;
    float alpha = (k == 0) ? 1.0f : 2.0f;
    float ang = 2.0f * PI_F * (float)(k * n) / (float)Nn;
    float v = (q < 16) ? cosf(ang) : -sinf(ang);
    Tinv[n * KC + q] = (_Float16)(v * alpha * inv);    // [128 n][32 q], K-contig
  }
}

// ---- K0b: pack complex mix weights, TRANSPOSED: W_T[k][col(256)][row(256)] ----
// logical W[k] = [[Wr, Wi], [-Wi, Wr]]   (w: [D,D,NM,2] = [i,o,k,ri])
__global__ void k_packw(const float* __restrict__ w, _Float16* __restrict__ WpT) {
  int idx = blockIdx.x * blockDim.x + threadIdx.x;
  if (idx >= NMODE * 256 * 256) return;
  int k = idx >> 16;
  int r = (idx >> 8) & 255;   // K index (input-channel block)
  int c = idx & 255;          // N index (output block)
  int i = r & 127, o = c & 127;
  float wr = w[((i * Dn + o) * NMODE + k) * 2 + 0];
  float wi = w[((i * Dn + o) * NMODE + k) * 2 + 1];
  float v = (r < 128) ? ((c < 128) ? wr : wi)
                      : ((c < 128) ? -wi : wr);
  WpT[((size_t)k * 256 + c) * 256 + r] = (_Float16)v;  // N-major, K-contig
}

// ---- K0c: f32 -> f16 transposing convert: d[c][rows] = s[r][c] ----
__global__ void k_cvt_t(const float* __restrict__ s, _Float16* __restrict__ d,
                        int rows, int cols) {
  int idx = blockIdx.x * blockDim.x + threadIdx.x;
  if (idx >= rows * cols) return;
  int r = idx / cols, c = idx % cols;
  d[(size_t)c * rows + r] = (_Float16)s[idx];
}

// ---- K1/K2: truncated forward DFT along y (xdir=0) or x (xdir=1) ----
// S[kc, ch] = Ftab @ Xslab ; X staged transposed in LDS: Xs[ch][n] (K-contig B)
// Output S[k][row][256] = [Sr(128) | Si(128)]  (A-operand of mix, K-contig)
__global__ void k_dft(const float* __restrict__ x, const _Float16* __restrict__ Ftab,
                      _Float16* __restrict__ S, int xdir) {
  __shared__ _Float16 Xs[Dn * XLD];   // ~34 KB, [ch][n]
  const int r = blockIdx.x;           // bm (y-dir) or bn (x-dir)
  if (!xdir) {
    const float4* src = (const float4*)(x + (size_t)r * Nn * Dn);
    for (int t = threadIdx.x; t < Nn * Dn / 4; t += blockDim.x) {
      float4 v = src[t];              // x[n][i..i+3]
      int n = t >> 5, i = (t & 31) * 4;
      Xs[(i + 0) * XLD + n] = (_Float16)v.x;
      Xs[(i + 1) * XLD + n] = (_Float16)v.y;
      Xs[(i + 2) * XLD + n] = (_Float16)v.z;
      Xs[(i + 3) * XLD + n] = (_Float16)v.w;
    }
  } else {
    const int b = r >> 7, n = r & 127;
    for (int t = threadIdx.x; t < Mn * Dn / 4; t += blockDim.x) {
      int m = t >> 5, i = (t & 31) * 4;
      float4 v = *(const float4*)(x + ((size_t)(b * Mn + m) * Nn + n) * Dn + i);
      Xs[(i + 0) * XLD + m] = (_Float16)v.x;
      Xs[(i + 1) * XLD + m] = (_Float16)v.y;
      Xs[(i + 2) * XLD + m] = (_Float16)v.z;
      Xs[(i + 3) * XLD + m] = (_Float16)v.w;
    }
  }
  __syncthreads();
  const int wid = threadIdx.x >> 5, lane = threadIdx.x & 31;
  const int rbl = (lane >> 4) << 3, cl = lane & 15;
#pragma unroll
  for (int t = 0; t < 2; ++t) {                 // 16 tiles (2x8) over 8 waves
    int id = wid * 2 + t;
    int tr = id >> 3, tc = id & 7;
    v8f acc = {};
#pragma unroll
    for (int kk = 0; kk < 4; ++kk) {
      v16h a = frag_k(Ftab + tr * 16 * Nn + kk * 32, Nn);
      v16h b = frag_k(Xs + tc * 16 * XLD + kk * 32, XLD);
      acc = wmma_f16(a, b, acc);
    }
#pragma unroll
    for (int j = 0; j < 8; ++j) {
      int kc = tr * 16 + rbl + j;
      int k = kc & 15, half = kc >> 4;
      int i = tc * 16 + cl;
      S[((size_t)k * ROWS + r) * 256 + half * 128 + i] = (_Float16)acc[j];
    }
  }
}

// ---- K3: per-mode complex channel mixing, k-batched GEMM [2048x256]@[256x256]
// A block staged in LDS by the Tensor Data Mover (32 KB contiguous f16).
__global__ void k_mix(const _Float16* __restrict__ Sy, const _Float16* __restrict__ Sx,
                      const _Float16* __restrict__ WyT, const _Float16* __restrict__ WxT,
                      _Float16* __restrict__ Ty, _Float16* __restrict__ Tx) {
  __shared__ _Float16 As[64 * 256];  // 32 KB
  const int rb0 = blockIdx.x * 64;
  const int k   = blockIdx.y;
  const int dir = blockIdx.z;
  const _Float16* Sm = (dir ? Sx : Sy) + (size_t)k * ROWS * 256;
  const _Float16* WT = (dir ? WxT : WyT) + (size_t)k * 256 * 256;
  _Float16* T = dir ? Tx : Ty;
#if HAVE_TDM
  if (threadIdx.x < 32) {            // wave 0 issues the DMA, waits TENSORcnt
    tdm_copy_1d(Sm + (size_t)rb0 * 256, As, 64 * 256 * 2 / 8);
  }
#else
  {
    const uint4* src = (const uint4*)(Sm + (size_t)rb0 * 256);
    uint4* dst = (uint4*)As;
    for (int t = threadIdx.x; t < 64 * 256 / 8; t += blockDim.x) dst[t] = src[t];
  }
#endif
  __builtin_prefetch(WT, 0, 1);      // global_prefetch_b8: warm L2 for mode weights
  __syncthreads();
  const int wid = threadIdx.x >> 5, lane = threadIdx.x & 31;
  const int rbl = (lane >> 4) << 3, cl = lane & 15;
#pragma unroll
  for (int t = 0; t < 8; ++t) {                 // 64 tiles (4x16) over 8 waves
    int id = wid * 8 + t;
    int tr = id >> 4, tc = id & 15;
    v8f acc = {};
#pragma unroll
    for (int kk = 0; kk < 8; ++kk) {
      v16h a = frag_k(As + tr * 16 * 256 + kk * 32, 256);
      v16h b = frag_k(WT + tc * 16 * 256 + kk * 32, 256);
      acc = wmma_f16(a, b, acc);
    }
#pragma unroll
    for (int j = 0; j < 8; ++j) {
      int row = rb0 + tr * 16 + rbl + j;
      int c = tc * 16 + cl;
      int q = (c < 128) ? k : (16 + k);
      T[((size_t)row * 128 + (c & 127)) * KC + q] = (_Float16)acc[j];
    }
  }
}

// ---- K4/K5: inverse DFT, write (y) then accumulate (x) into h [B,M,N,D] ----
// out[s, o] = Tinv[128x32] @ T[r] ; T staged as Us[o][q] via TDM (8 KB f16)
__global__ void k_inv(const _Float16* __restrict__ T, const _Float16* __restrict__ Tinv,
                      float* __restrict__ h, int xdir) {
  __shared__ _Float16 Us[Dn * KC];   // 8 KB, [o][q]
  const int r = blockIdx.x;          // bm (y) or bn (x)
#if HAVE_TDM
  if (threadIdx.x < 32) {
    tdm_copy_1d(T + (size_t)r * Dn * KC, Us, Dn * KC * 2 / 8);
  }
#else
  {
    const uint4* src = (const uint4*)(T + (size_t)r * Dn * KC);
    uint4* dst = (uint4*)Us;
    for (int t = threadIdx.x; t < Dn * KC / 8; t += blockDim.x) dst[t] = src[t];
  }
#endif
  __syncthreads();
  const int wid = threadIdx.x >> 5, lane = threadIdx.x & 31;
  const int rbl = (lane >> 4) << 3, cl = lane & 15;
#pragma unroll
  for (int t = 0; t < 8; ++t) {                 // 64 tiles (8x8) over 8 waves
    int id = wid * 8 + t;
    int tr = id >> 3, tc = id & 7;
    v16h a = frag_k(Tinv + tr * 16 * KC, KC);   // K=32: single WMMA per tile
    v16h b = frag_k(Us + tc * 16 * KC, KC);
    v8f acc = {};
    acc = wmma_f16(a, b, acc);
#pragma unroll
    for (int j = 0; j < 8; ++j) {
      int s = tr * 16 + rbl + j;                // n (y-dir) or m (x-dir)
      int o = tc * 16 + cl;
      if (!xdir) {
        h[((size_t)r * Nn + s) * Dn + o] = acc[j];
      } else {
        int b_ = r >> 7, n = r & 127;
        size_t idx = ((size_t)(b_ * Mn + s) * Nn + n) * Dn + o;
        h[idx] += acc[j];                       // unique owner per element
      }
    }
  }
}

// ---- K6: fused FFN (128->512 ReLU 512->128) + LayerNorm, 32 rows/block ----
// Weights consumed transposed (K-contig): W0T[512][128], W1T[128][512]
__global__ void k_ffn(const float* __restrict__ h,
                      const _Float16* __restrict__ W0T, const float* __restrict__ b0,
                      const _Float16* __restrict__ W1T, const float* __restrict__ b1,
                      const float* __restrict__ g, const float* __restrict__ be,
                      float* __restrict__ out) {
  __shared__ _Float16 As[32 * Dn];   // 8 KB   [row][k]   K-contig
  __shared__ _Float16 Hs[32 * Hn];   // 32 KB  [row][h]   K-contig for GEMM2 A
  __shared__ float    Os[32 * Dn];   // 16 KB
  const size_t row0 = (size_t)blockIdx.x * 32;
  const float4* src = (const float4*)(h + row0 * Dn);
  for (int t = threadIdx.x; t < 32 * Dn / 4; t += blockDim.x) {
    float4 v = src[t];
    _Float16* d = &As[t * 4];
    d[0] = (_Float16)v.x; d[1] = (_Float16)v.y;
    d[2] = (_Float16)v.z; d[3] = (_Float16)v.w;
  }
  __syncthreads();
  const int wid = threadIdx.x >> 5, lane = threadIdx.x & 31;
  const int rbl = (lane >> 4) << 3, cl = lane & 15;
  // GEMM1: 32x512, K=128; tiles 2x32 = 64 over 8 waves
#pragma unroll
  for (int t = 0; t < 8; ++t) {
    int id = wid * 8 + t;
    int tr = id >> 5, tc = id & 31;
    v8f acc = {};
#pragma unroll
    for (int kk = 0; kk < 4; ++kk) {
      v16h a = frag_k(As + tr * 16 * Dn + kk * 32, Dn);
      v16h b = frag_k(W0T + tc * 16 * Dn + kk * 32, Dn);
      acc = wmma_f16(a, b, acc);
    }
#pragma unroll
    for (int j = 0; j < 8; ++j) {
      int row = tr * 16 + rbl + j;
      int c = tc * 16 + cl;
      float v = acc[j] + b0[c];
      Hs[row * Hn + c] = (_Float16)(v > 0.0f ? v : 0.0f);
    }
  }
  __syncthreads();
  // GEMM2: 32x128, K=512; tiles 2x8 = 16 over 8 waves
#pragma unroll
  for (int t = 0; t < 2; ++t) {
    int id = wid * 2 + t;
    int tr = id >> 3, tc = id & 7;
    v8f acc = {};
#pragma unroll
    for (int kk = 0; kk < 16; ++kk) {
      v16h a = frag_k(Hs + tr * 16 * Hn + kk * 32, Hn);
      v16h b = frag_k(W1T + tc * 16 * Hn + kk * 32, Hn);
      acc = wmma_f16(a, b, acc);
    }
#pragma unroll
    for (int j = 0; j < 8; ++j) {
      int row = tr * 16 + rbl + j;
      int c = tc * 16 + cl;
      Os[row * Dn + c] = acc[j] + b1[c];
    }
  }
  __syncthreads();
  // LayerNorm: each wave normalizes 4 rows (32 lanes x 4 elems = 128)
  for (int rr = 0; rr < 4; ++rr) {
    int row = wid * 4 + rr;
    float v[4]; float s = 0.0f, sq = 0.0f;
#pragma unroll
    for (int e = 0; e < 4; ++e) {
      v[e] = Os[row * Dn + lane * 4 + e];
      s += v[e]; sq += v[e] * v[e];
    }
#pragma unroll
    for (int m = 16; m >= 1; m >>= 1) {
      s  += __shfl_xor(s, m, 32);
      sq += __shfl_xor(sq, m, 32);
    }
    float mean = s * (1.0f / 128.0f);
    float var  = sq * (1.0f / 128.0f) - mean * mean;
    float rstd = rsqrtf(var + LN_EPS);
#pragma unroll
    for (int e = 0; e < 4; ++e) {
      int c = lane * 4 + e;
      out[(row0 + row) * Dn + c] = (v[e] - mean) * rstd * g[c] + be[c];
    }
  }
}

extern "C" void kernel_launch(void* const* d_in, const int* in_sizes, int n_in,
                              void* d_out, int out_size, void* d_ws, size_t ws_size,
                              hipStream_t stream) {
  const float* x    = (const float*)d_in[0];
  const float* w1   = (const float*)d_in[1];
  const float* w2   = (const float*)d_in[2];
  const float* ffw0 = (const float*)d_in[3];
  const float* ffb0 = (const float*)d_in[4];
  const float* ffw1 = (const float*)d_in[5];
  const float* ffb1 = (const float*)d_in[6];
  const float* ln_g = (const float*)d_in[7];
  const float* ln_b = (const float*)d_in[8];
  float* out = (float*)d_out;

  // workspace carve-up (256B aligned)
  char* base = (char*)d_ws;
  size_t off = 0;
  auto carve = [&](size_t bytes) -> void* {
    void* p = base + off;
    off = (off + bytes + 255) & ~(size_t)255;
    return p;
  };
  _Float16* Ftab = (_Float16*)carve(KC * Nn * 2);
  _Float16* Tinv = (_Float16*)carve(Nn * KC * 2);
  _Float16* WyT  = (_Float16*)carve((size_t)NMODE * 256 * 256 * 2);
  _Float16* WxT  = (_Float16*)carve((size_t)NMODE * 256 * 256 * 2);
  _Float16* W0T  = (_Float16*)carve((size_t)Dn * Hn * 2);
  _Float16* W1T  = (_Float16*)carve((size_t)Hn * Dn * 2);
  _Float16* Sy   = (_Float16*)carve((size_t)NMODE * ROWS * 256 * 2);
  _Float16* Sx   = (_Float16*)carve((size_t)NMODE * ROWS * 256 * 2);
  _Float16* Ty   = (_Float16*)carve((size_t)ROWS * Dn * KC * 2);
  _Float16* Tx   = (_Float16*)carve((size_t)ROWS * Dn * KC * 2);
  float*    hbuf = (float*)carve((size_t)Bn * Mn * Nn * Dn * 4);
  (void)ws_size; (void)in_sizes; (void)n_in; (void)out_size;

  // init: tables + weight packing/conversion (transposed, K-contiguous)
  k_tables<<<1, 256, 0, stream>>>(Ftab, Tinv);
  int npack = NMODE * 256 * 256;
  k_packw<<<(npack + 255) / 256, 256, 0, stream>>>(w1, WyT);
  k_packw<<<(npack + 255) / 256, 256, 0, stream>>>(w2, WxT);
  k_cvt_t<<<(Dn * Hn + 255) / 256, 256, 0, stream>>>(ffw0, W0T, Dn, Hn);
  k_cvt_t<<<(Hn * Dn + 255) / 256, 256, 0, stream>>>(ffw1, W1T, Hn, Dn);

  // Fourier pipeline (all WMMA GEMMs; TDM staging where data is already f16)
  k_dft<<<ROWS, 256, 0, stream>>>(x, Ftab, Sy, 0);
  k_dft<<<ROWS, 256, 0, stream>>>(x, Ftab, Sx, 1);
  k_mix<<<dim3(ROWS / 64, NMODE, 2), 256, 0, stream>>>(Sy, Sx, WyT, WxT, Ty, Tx);
  k_inv<<<ROWS, 256, 0, stream>>>(Ty, Tinv, hbuf, 0);
  k_inv<<<ROWS, 256, 0, stream>>>(Tx, Tinv, hbuf, 1);

  // fused FFN + LayerNorm
  k_ffn<<<(Bn * Mn * Nn) / 32, 256, 0, stream>>>(hbuf, W0T, ffb0, W1T, ffb1,
                                                 ln_g, ln_b, out);
}
